// PowerAmpStage_28810640621782
// MI455X (gfx1250) — compile-verified, matrix operand order checked by reading, not run on hardware
//
#include <hip/hip_runtime.h>
#include <hip/hip_bf16.h>

// ---------------------------------------------------------------- constants
#define B_      8
#define L_      131072
#define NL_     (B_ * L_)
#define CHUNK   256            // affine-scan chunk length (power of 2)
#define NCH     (L_ / CHUNK)   // 512 chunks per row
#define NBLK    (L_ / 16)      // 8192 16-sample blocks per row (envelope)
#define NBLK_T  (B_ * NBLK)    // 65536
#define TILE_G  512            // GRU LDS tile (per row)

// parameter-block float offsets (in d_ws)
constexpr int P_C        = 0;
constexpr int P_C16      = 1;
constexpr int P_FBMIX    = 2;
constexpr int P_CPOW     = 4;    // 16: c^0..c^15
constexpr int P_DEPTH    = 20;   // 8
constexpr int P_APRE     = 32;   // 16
constexpr int P_APRE_CH  = 48;   // 16 : A^CHUNK
constexpr int P_APOST    = 64;   // 16
constexpr int P_APOST_CH = 80;   // 16
constexpr int P_KPRE     = 96;   // 4
constexpr int P_KPOST    = 100;  // 4
constexpr int P_GPRE     = 104;  // 2 : y = g0*x + g1*s1a + s1b
constexpr int P_GPOST    = 106;  // 2
constexpr int P_FBA      = 112;  // 8 rows * 4 (2x2 A per row)
constexpr int P_FBA_CH   = 144;  // 8 rows * 4 (A^CHUNK per row)
constexpr int P_KFB      = 176;  // 8 rows * 2
constexpr int P_FBB0     = 192;  // 8
constexpr int P_HF       = 256;  // 32 lanes * 8 : precomputed WMMA A-fragments of H
constexpr int PARAM_FLOATS = 1024;

typedef float v2f_t  __attribute__((ext_vector_type(2)));
typedef float v4f_t  __attribute__((ext_vector_type(4)));
typedef float v8f_t  __attribute__((ext_vector_type(8)));
typedef _Float16 v16h_t __attribute__((ext_vector_type(16)));
typedef unsigned int v4u_t __attribute__((ext_vector_type(4)));
typedef int v8i_t __attribute__((ext_vector_type(8)));
typedef int v4i_t __attribute__((ext_vector_type(4)));

#if __has_builtin(__builtin_amdgcn_tensor_load_to_lds) && \
    __has_builtin(__builtin_amdgcn_tensor_store_from_lds) && \
    __has_builtin(__builtin_amdgcn_s_wait_tensorcnt)
#define HAVE_TDM 1
#else
#define HAVE_TDM 0
#endif

// ---------------------------------------------------------------- helpers
__device__ __forceinline__ float fast_sigmoid(float v) {
    return 1.0f / (1.0f + __expf(-v));
}
__device__ __forceinline__ float fast_tanh(float v) {
    float a = fabsf(v);
    float e = __expf(-2.0f * a);
    float t = (1.0f - e) / (1.0f + e);
    return copysignf(t, v);
}

__device__ void mat4mul(const float* X, const float* Y, float* Z) {
    for (int i = 0; i < 4; ++i)
        for (int j = 0; j < 4; ++j) {
            float s = 0.0f;
            for (int k = 0; k < 4; ++k) s += X[i * 4 + k] * Y[k * 4 + j];
            Z[i * 4 + j] = s;
        }
}
__device__ void mat4_pow_chunk(float* A) {   // A <- A^(2^8) = A^CHUNK
    float T[16];
    for (int s = 0; s < 8; ++s) {
        mat4mul(A, A, T);
        for (int i = 0; i < 16; ++i) A[i] = T[i];
    }
}
__device__ void mat2_pow_chunk(float* A) {
    for (int s = 0; s < 8; ++s) {
        float t0 = A[0]*A[0] + A[1]*A[2];
        float t1 = A[0]*A[1] + A[1]*A[3];
        float t2 = A[2]*A[0] + A[3]*A[2];
        float t3 = A[2]*A[1] + A[3]*A[3];
        A[0]=t0; A[1]=t1; A[2]=t2; A[3]=t3;
    }
}

// _stable(): raw[5] -> {b0,b1,b2,a1,a2}
__device__ void stable5(const float* raw, float* o) {
    o[0] = raw[0]; o[1] = raw[1]; o[2] = raw[2];
    float a1 = 2.0f * fast_tanh(raw[3]);
    float a2 = ((2.0f - fabsf(a1)) * fast_tanh(raw[4]) + fabsf(a1)) * 0.5f;
    o[3] = a1; o[4] = a2;
}

// two-biquad cascade (TDF-II) -> 4-dim affine recurrence s' = A s + k x,
// output y = g0*x + g1*s1a + s1b
__device__ void build_cascade(const float* sA, const float* sB,
                              float* A, float* k, float* g) {
    float b0a=sA[0], b1a=sA[1], b2a=sA[2], a1a=sA[3], a2a=sA[4];
    float b0b=sB[0], b1b=sB[1], b2b=sB[2], a1b=sB[3], a2b=sB[4];
    float c1a = b1a - a1a*b0a, c2a = b2a - a2a*b0a;
    float c1b = b1b - a1b*b0b, c2b = b2b - a2b*b0b;
    for (int i = 0; i < 16; ++i) A[i] = 0.0f;
    A[0]  = -a1a; A[1]  = 1.0f;
    A[4]  = -a2a;
    A[8]  =  c1b; A[10] = -a1b; A[11] = 1.0f;
    A[12] =  c2b; A[14] = -a2b;
    k[0] = c1a; k[1] = c2a; k[2] = c1b * b0a; k[3] = c2b * b0a;
    g[0] = b0a * b0b; g[1] = b0b;
}

#if HAVE_TDM
// Build a TDM D# (groups 0/1) for a 2D tile: B_ rows x TILE_G f32 elements,
// row stride L_ elements, per CDNA5 ISA bit layout.
__device__ __forceinline__ void tdm_desc_2d(unsigned long long gaddr, unsigned int lds_off,
                                            v4u_t& g0, v8i_t& g1) {
    g0[0] = 1u;                                            // count=1, user descriptor
    g0[1] = lds_off;                                       // lds_addr (bytes)
    g0[2] = (unsigned int)(gaddr & 0xffffffffull);         // global_addr[31:0]
    g0[3] = (unsigned int)((gaddr >> 32) & 0x1ffffffull)   // global_addr[56:32]
          | (2u << 30);                                    // type=2 ("image")
    g1[0] = (int)(2u << 16);                 // workgroup_mask=0, data_size=2 (4B)
    g1[1] = (int)((L_ & 0xffffu) << 16);     // tensor_dim0[15:0] -> bits 63:48
    g1[2] = (int)((L_ >> 16) | (B_ << 16));  // tensor_dim0[31:16] | tensor_dim1[15:0]
    g1[3] = (int)(TILE_G << 16);             // tensor_dim1[31:16]=0 | tile_dim0
    g1[4] = (int)B_;                         // tile_dim1 | tile_dim2=0
    g1[5] = (int)L_;                         // tensor_dim0_stride[31:0]
    g1[6] = 0;                               // stride0[47:32] | stride1[15:0]
    g1[7] = 0;                               // tensor_dim1_stride[47:16]
}
#endif

// ---------------------------------------------------------------- kernel 0: params
__global__ void param_kernel(const float* __restrict__ knobs,
                             const float* __restrict__ env_raw,
                             const float* __restrict__ pre_raw,
                             const float* __restrict__ post_raw,
                             const float* __restrict__ sw1, const float* __restrict__ sb1,
                             const float* __restrict__ sw2, const float* __restrict__ sb2,
                             const float* __restrict__ fw1, const float* __restrict__ fb1,
                             const float* __restrict__ fw2, const float* __restrict__ fb2,
                             const float* __restrict__ famt,
                             float* __restrict__ P) {
    if (threadIdx.x != 0 || blockIdx.x != 0) return;

    float c = fast_sigmoid(env_raw[0]);
    P[P_C] = c;
    float cp = 1.0f;
    for (int i = 0; i < 16; ++i) { P[P_CPOW + i] = cp; cp *= c; }
    P[P_C16]   = cp;                       // c^16
    P[P_FBMIX] = fast_sigmoid(famt[0]);

    // precompute per-lane WMMA A-fragments of H (16x16 lower-tri Toeplitz):
    // lane l, slice q: a.x = H[m][k0], a.y = H[m][k1], k0 = 4q + (l>=16 ? 2:0)
    float omc = 1.0f - c;
    for (int l = 0; l < 32; ++l) {
        int m = l & 15;
        int koff = (l >= 16) ? 2 : 0;
        for (int q = 0; q < 4; ++q) {
            int k0 = 4 * q + koff, k1 = k0 + 1;
            P[P_HF + l * 8 + q * 2 + 0] = (m >= k0) ? omc * P[P_CPOW + (m - k0)] : 0.0f;
            P[P_HF + l * 8 + q * 2 + 1] = (m >= k1) ? omc * P[P_CPOW + (m - k1)] : 0.0f;
        }
    }

    // sag depth per row (1 -> 8 -> 1 MLP)
    for (int r = 0; r < B_; ++r) {
        float kn = knobs[r * 3 + 0];
        float acc = sb2[0];
        for (int j = 0; j < 8; ++j)
            acc += fast_tanh(kn * sw1[j] + sb1[j]) * sw2[j];
        P[P_DEPTH + r] = fast_sigmoid(acc);
    }

    // pre / post cascades
    float S[10], A4[16], k4[4], g2[2], M[16];
    stable5(pre_raw,     S);
    stable5(pre_raw + 5, S + 5);
    build_cascade(S, S + 5, A4, k4, g2);
    for (int i = 0; i < 16; ++i) P[P_APRE + i] = A4[i];
    for (int i = 0; i < 4;  ++i) P[P_KPRE + i] = k4[i];
    P[P_GPRE] = g2[0]; P[P_GPRE + 1] = g2[1];
    for (int i = 0; i < 16; ++i) M[i] = A4[i];
    mat4_pow_chunk(M);
    for (int i = 0; i < 16; ++i) P[P_APRE_CH + i] = M[i];

    stable5(post_raw,     S);
    stable5(post_raw + 5, S + 5);
    build_cascade(S, S + 5, A4, k4, g2);
    for (int i = 0; i < 16; ++i) P[P_APOST + i] = A4[i];
    for (int i = 0; i < 4;  ++i) P[P_KPOST + i] = k4[i];
    P[P_GPOST] = g2[0]; P[P_GPOST + 1] = g2[1];
    for (int i = 0; i < 16; ++i) M[i] = A4[i];
    mat4_pow_chunk(M);
    for (int i = 0; i < 16; ++i) P[P_APOST_CH + i] = M[i];

    // per-row feedback biquad from knobs MLP (2 -> 16 -> 5) -> _stable
    for (int r = 0; r < B_; ++r) {
        float k1 = knobs[r * 3 + 1], k2 = knobs[r * 3 + 2];
        float h[16];
        for (int j = 0; j < 16; ++j)
            h[j] = fast_tanh(k1 * fw1[j * 2 + 0] + k2 * fw1[j * 2 + 1] + fb1[j]);
        float raw[5];
        for (int i = 0; i < 5; ++i) {
            float s = fb2[i];
            for (int j = 0; j < 16; ++j) s += h[j] * fw2[i * 16 + j];
            raw[i] = s;
        }
        float cf[5];
        stable5(raw, cf);
        float b0 = cf[0], b1c = cf[1], b2c = cf[2], a1 = cf[3], a2 = cf[4];
        float A2[4] = { -a1, 1.0f, -a2, 0.0f };
        P[P_FBA + r * 4 + 0] = A2[0]; P[P_FBA + r * 4 + 1] = A2[1];
        P[P_FBA + r * 4 + 2] = A2[2]; P[P_FBA + r * 4 + 3] = A2[3];
        P[P_KFB + r * 2 + 0] = b1c - a1 * b0;
        P[P_KFB + r * 2 + 1] = b2c - a2 * b0;
        P[P_FBB0 + r] = b0;
        mat2_pow_chunk(A2);
        for (int i = 0; i < 4; ++i) P[P_FBA_CH + r * 4 + i] = A2[i];
    }
}

// ---------------------------------------------------------------- kernel 1: envelope local pass via WMMA
// E_local(block)[i] = sum_{k<=i} (1-c) c^(i-k) |x[k]|  ==  H(16x16) * |X|(16xN)
__global__ void __launch_bounds__(256)
env_local_kernel(const float* __restrict__ x, const float* __restrict__ P,
                 float* __restrict__ elocal, float* __restrict__ lastlocal) {
    const int lane = threadIdx.x & 31;
    const int wave = blockIdx.x * (blockDim.x >> 5) + (threadIdx.x >> 5);
    const int tilesPerRow = L_ / 256;         // 512
    const int row  = wave / tilesPerRow;      // grid sized exactly: EXEC all-1 for WMMA
    const int tcol = wave % tilesPerRow;

    const int  m  = lane & 15;
    const bool hi = lane >= 16;
    const size_t base = (size_t)row * L_ + (size_t)tcol * 256;

    v8f_t acc = {0.f, 0.f, 0.f, 0.f, 0.f, 0.f, 0.f, 0.f};

#if __has_builtin(__builtin_amdgcn_wmma_f32_16x16x4_f32)
    // precomputed A-fragments: 8 floats per lane (two b128 loads)
    const v4f_t* hf = reinterpret_cast<const v4f_t*>(P + P_HF + lane * 8);
    const v4f_t h0 = hf[0], h1 = hf[1];
    // B loads at immediate offsets from a per-lane base
    const float* xs = x + base + m * 16 + (hi ? 2 : 0);

    v2f_t a, b;
    a.x = h0.x; a.y = h0.y;
    b.x = fabsf(xs[0]);  b.y = fabsf(xs[1]);
    acc = __builtin_amdgcn_wmma_f32_16x16x4_f32(false, a, false, b, (short)0, acc, false, false);
    a.x = h0.z; a.y = h0.w;
    b.x = fabsf(xs[4]);  b.y = fabsf(xs[5]);
    acc = __builtin_amdgcn_wmma_f32_16x16x4_f32(false, a, false, b, (short)0, acc, false, false);
    a.x = h1.x; a.y = h1.y;
    b.x = fabsf(xs[8]);  b.y = fabsf(xs[9]);
    acc = __builtin_amdgcn_wmma_f32_16x16x4_f32(false, a, false, b, (short)0, acc, false, false);
    a.x = h1.z; a.y = h1.w;
    b.x = fabsf(xs[12]); b.y = fabsf(xs[13]);
    acc = __builtin_amdgcn_wmma_f32_16x16x4_f32(false, a, false, b, (short)0, acc, false, false);
#else
    // probe-confirmed fallback: f16 inputs, K padded 16->32 with zeros
    const float c   = P[P_C];
    const float omc = 1.0f - c;
    float cpw[16];
#pragma unroll
    for (int i = 0; i < 16; ++i) cpw[i] = P[P_CPOW + i];
    v16h_t a, b;
#pragma unroll
    for (int j = 0; j < 16; ++j) {
        int k = hi ? ((j < 8) ? j + 8 : j + 16) : ((j < 8) ? j : j + 8);
        float av = (k < 16 && m >= k) ? omc * cpw[m - k] : 0.0f;
        float bv = (k < 16) ? fabsf(x[base + m * 16 + k]) : 0.0f;
        a[j] = (_Float16)av;
        b[j] = (_Float16)bv;
    }
    acc = __builtin_amdgcn_wmma_f32_16x16x32_f16(false, a, false, b, (short)0, acc, false, false);
#endif

    // C/D layout: lane<16 holds rows 0..7 of column m, lane>=16 rows 8..15 of column m
    const int i0 = hi ? 8 : 0;
    float* dst = elocal + base + m * 16 + i0;
    v4f_t lo = {acc[0], acc[1], acc[2], acc[3]};
    v4f_t h4 = {acc[4], acc[5], acc[6], acc[7]};
    *reinterpret_cast<v4f_t*>(dst)     = lo;
    *reinterpret_cast<v4f_t*>(dst + 4) = h4;
    if (hi) lastlocal[row * NBLK + tcol * 16 + m] = acc[7];   // E_local[15]
}

// ---------------------------------------------------------------- kernel 2: envelope block carry scan
__global__ void env_carry_kernel(const float* __restrict__ P,
                                 const float* __restrict__ ll,
                                 float* __restrict__ Ein) {
    int r = threadIdx.x;
    if (r >= B_) return;
    const float c16 = P[P_C16];
    const float* src = ll  + (size_t)r * NBLK;
    float*       dst = Ein + (size_t)r * NBLK;
    float e = 0.0f;
    for (int b = 0; b < NBLK; ++b) {
        if (b + 64 < NBLK) __builtin_prefetch(src + b + 64, 0, 0);  // global_prefetch_b8
        dst[b] = e;
        e = fmaf(c16, e, src[b]);
    }
}

// ---------------------------------------------------------------- kernel 3: apply envelope + sag gain
__global__ void __launch_bounds__(256)
sag_apply_kernel(const float* __restrict__ x, const float* __restrict__ P,
                 const float* __restrict__ elocal, const float* __restrict__ Ein,
                 float* __restrict__ y1) {
    size_t idx = (size_t)blockIdx.x * blockDim.x + threadIdx.x;   // exact grid NL_
    int row = (int)(idx / L_);
    int t   = (int)(idx % L_);
    int i   = t & 15;
    int blk = t >> 4;
    float e = elocal[idx] + P[P_CPOW + i] * P[P_C] * Ein[(size_t)row * NBLK + blk];
    y1[idx] = x[idx] * (1.0f - P[P_DEPTH + row] * e);
}

// ---------------------------------------------------------------- cascade (4-state) scan kernels
__global__ void __launch_bounds__(256)
cascade_w_kernel(const float* __restrict__ in, const float* __restrict__ A,
                 const float* __restrict__ kv, float* __restrict__ w) {
    int tid = blockIdx.x * blockDim.x + threadIdx.x;   // B_*NCH = 4096
    int row = tid / NCH, ch = tid % NCH;
    const float* src = in + (size_t)row * L_ + (size_t)ch * CHUNK;
    float a[16];
#pragma unroll
    for (int i = 0; i < 16; ++i) a[i] = A[i];
    float k0 = kv[0], k1 = kv[1], k2 = kv[2], k3 = kv[3];
    float v0 = 0, v1 = 0, v2 = 0, v3 = 0;
    for (int i = 0; i < CHUNK; ++i) {
        float xv = src[i];
        float n0 = a[0]*v0 + a[1]*v1 + a[2]*v2 + a[3]*v3 + k0 * xv;
        float n1 = a[4]*v0 + a[5]*v1 + a[6]*v2 + a[7]*v3 + k1 * xv;
        float n2 = a[8]*v0 + a[9]*v1 + a[10]*v2 + a[11]*v3 + k2 * xv;
        float n3 = a[12]*v0 + a[13]*v1 + a[14]*v2 + a[15]*v3 + k3 * xv;
        v0 = n0; v1 = n1; v2 = n2; v3 = n3;
    }
    float* o = w + (size_t)tid * 4;
    o[0] = v0; o[1] = v1; o[2] = v2; o[3] = v3;
}

__global__ void cascade_carry_kernel(const float* __restrict__ Ach,
                                     const float* __restrict__ w,
                                     float* __restrict__ sin) {
    int r = threadIdx.x;
    if (r >= B_) return;
    float a[16];
#pragma unroll
    for (int i = 0; i < 16; ++i) a[i] = Ach[i];
    float v0 = 0, v1 = 0, v2 = 0, v3 = 0;
    for (int j = 0; j < NCH; ++j) {
        size_t b = ((size_t)r * NCH + j) * 4;
        if (j + 16 < NCH) __builtin_prefetch(w + b + 64, 0, 0);
        sin[b] = v0; sin[b + 1] = v1; sin[b + 2] = v2; sin[b + 3] = v3;
        float w0 = w[b], w1 = w[b + 1], w2 = w[b + 2], w3 = w[b + 3];
        float n0 = a[0]*v0 + a[1]*v1 + a[2]*v2 + a[3]*v3 + w0;
        float n1 = a[4]*v0 + a[5]*v1 + a[6]*v2 + a[7]*v3 + w1;
        float n2 = a[8]*v0 + a[9]*v1 + a[10]*v2 + a[11]*v3 + w2;
        float n3 = a[12]*v0 + a[13]*v1 + a[14]*v2 + a[15]*v3 + w3;
        v0 = n0; v1 = n1; v2 = n2; v3 = n3;
    }
}

__global__ void __launch_bounds__(256)
cascade_replay_kernel(const float* __restrict__ in, const float* __restrict__ A,
                      const float* __restrict__ kv, const float* __restrict__ g,
                      const float* __restrict__ sin, float* __restrict__ out) {
    int tid = blockIdx.x * blockDim.x + threadIdx.x;
    int row = tid / NCH, ch = tid % NCH;
    const float* src = in  + (size_t)row * L_ + (size_t)ch * CHUNK;
    float*       dst = out + (size_t)row * L_ + (size_t)ch * CHUNK;
    float a[16];
#pragma unroll
    for (int i = 0; i < 16; ++i) a[i] = A[i];
    float k0 = kv[0], k1 = kv[1], k2 = kv[2], k3 = kv[3];
    float g0 = g[0], gb = g[1];
    size_t b = (size_t)tid * 4;
    float v0 = sin[b], v1 = sin[b + 1], v2 = sin[b + 2], v3 = sin[b + 3];
    for (int i = 0; i < CHUNK; ++i) {
        float xv = src[i];
        dst[i] = g0 * xv + gb * v0 + v2;                 // cascade output (pre-update state)
        float n0 = a[0]*v0 + a[1]*v1 + a[2]*v2 + a[3]*v3 + k0 * xv;
        float n1 = a[4]*v0 + a[5]*v1 + a[6]*v2 + a[7]*v3 + k1 * xv;
        float n2 = a[8]*v0 + a[9]*v1 + a[10]*v2 + a[11]*v3 + k2 * xv;
        float n3 = a[12]*v0 + a[13]*v1 + a[14]*v2 + a[15]*v3 + k3 * xv;
        v0 = n0; v1 = n1; v2 = n2; v3 = n3;
    }
}

// ---------------------------------------------------------------- GRU (hidden=1): sequential, 8 rows on 8 lanes.
// Tiles staged LDS<->global with the Tensor Data Mover (double-buffered).
__global__ void __launch_bounds__(256)
gru_kernel(const float* __restrict__ in,
           const float* __restrict__ wi, const float* __restrict__ wh,
           const float* __restrict__ bi, const float* __restrict__ bh,
           const float* __restrict__ ow, const float* __restrict__ ob,
           float* __restrict__ out) {
    __shared__ float tile[2][B_][TILE_G];
    const int tid = threadIdx.x;
    const float wi0 = wi[0], wi1 = wi[1], wi2 = wi[2];
    const float wh0 = wh[0], wh1 = wh[1], wh2 = wh[2];
    const float bi0 = bi[0], bi1 = bi[1], bi2 = bi[2];
    const float bh0 = bh[0], bh1 = bh[1], bh2 = bh[2];
    const float owv = ow[0], obv = ob[0];
    const int NT = L_ / TILE_G;
    float h = 0.0f;

#if HAVE_TDM
    const unsigned int lds0 = (unsigned int)(uintptr_t)(&tile[0][0][0]);
    const unsigned int ldsBytes = B_ * TILE_G * 4;
    if (tid < 32) {   // wave 0 owns the TDM queue (TENSORcnt is per-wave)
        v4u_t g0; v8i_t g1; v4i_t gz4 = {0, 0, 0, 0};
        v8i_t gz8 = {0, 0, 0, 0, 0, 0, 0, 0};
        tdm_desc_2d((unsigned long long)(uintptr_t)in, lds0, g0, g1);
        __builtin_amdgcn_tensor_load_to_lds(g0, g1, gz4, gz4, gz8, 0);
    }
    for (int ti = 0; ti < NT; ++ti) {
        const int cur = ti & 1;
        if (tid < 32) __builtin_amdgcn_s_wait_tensorcnt(0);  // tile ready; old store retired
        __syncthreads();
        if (tid < 32 && (ti + 1) < NT) {                      // prefetch next tile via TDM
            v4u_t g0; v8i_t g1; v4i_t gz4 = {0, 0, 0, 0};
            v8i_t gz8 = {0, 0, 0, 0, 0, 0, 0, 0};
            tdm_desc_2d((unsigned long long)(uintptr_t)(in + (size_t)(ti + 1) * TILE_G),
                        lds0 + (unsigned int)((ti + 1) & 1) * ldsBytes, g0, g1);
            __builtin_amdgcn_tensor_load_to_lds(g0, g1, gz4, gz4, gz8, 0);
        }
        if (tid < B_) {
            const int r = tid;
            for (int tt = 0; tt < TILE_G; ++tt) {
                float xt  = tile[cur][r][tt];
                float gi0 = fmaf(xt, wi0, bi0), gi1 = fmaf(xt, wi1, bi1), gi2 = fmaf(xt, wi2, bi2);
                float gh0 = fmaf(h, wh0, bh0),  gh1 = fmaf(h, wh1, bh1),  gh2 = fmaf(h, wh2, bh2);
                float rr  = fast_sigmoid(gi0 + gh0);
                float zz  = fast_sigmoid(gi1 + gh1);
                float nn  = fast_tanh(fmaf(rr, gh2, gi2));
                h = fmaf(zz, h - nn, nn);                 // (1-z)n + z h
                tile[cur][r][tt] = fmaf(h, owv, obv);
            }
        }
        __syncthreads();
        if (tid < 32) {                                      // write results back via TDM
            v4u_t g0; v8i_t g1; v4i_t gz4 = {0, 0, 0, 0};
            v8i_t gz8 = {0, 0, 0, 0, 0, 0, 0, 0};
            tdm_desc_2d((unsigned long long)(uintptr_t)(out + (size_t)ti * TILE_G),
                        lds0 + (unsigned int)cur * ldsBytes, g0, g1);
            __builtin_amdgcn_tensor_store_from_lds(g0, g1, gz4, gz4, gz8, 0);
        }
    }
    if (tid < 32) __builtin_amdgcn_s_wait_tensorcnt(0);
#else
    for (int t0 = 0; t0 < L_; t0 += TILE_G) {
        for (int j = tid; j < B_ * TILE_G; j += 256) {
            int r = j / TILE_G, tt = j % TILE_G;
            tile[0][r][tt] = in[(size_t)r * L_ + t0 + tt];
            if (t0 + TILE_G < L_)
                __builtin_prefetch(&in[(size_t)r * L_ + t0 + TILE_G + tt], 0, 0);
        }
        __syncthreads();
        if (tid < B_) {
            const int r = tid;
            for (int tt = 0; tt < TILE_G; ++tt) {
                float xt  = tile[0][r][tt];
                float gi0 = fmaf(xt, wi0, bi0), gi1 = fmaf(xt, wi1, bi1), gi2 = fmaf(xt, wi2, bi2);
                float gh0 = fmaf(h, wh0, bh0),  gh1 = fmaf(h, wh1, bh1),  gh2 = fmaf(h, wh2, bh2);
                float rr  = fast_sigmoid(gi0 + gh0);
                float zz  = fast_sigmoid(gi1 + gh1);
                float nn  = fast_tanh(fmaf(rr, gh2, gi2));
                h = fmaf(zz, h - nn, nn);
                tile[0][r][tt] = fmaf(h, owv, obv);
            }
        }
        __syncthreads();
        for (int j = tid; j < B_ * TILE_G; j += 256) {
            int r = j / TILE_G, tt = j % TILE_G;
            out[(size_t)r * L_ + t0 + tt] = tile[0][r][tt];
        }
        __syncthreads();
    }
#endif
}

// ---------------------------------------------------------------- feedback biquad (2-state, per-row coefs)
__global__ void __launch_bounds__(256)
fb_w_kernel(const float* __restrict__ in, const float* __restrict__ P,
            float* __restrict__ w) {
    int tid = blockIdx.x * blockDim.x + threadIdx.x;
    int row = tid / NCH, ch = tid % NCH;
    const float* src = in + (size_t)row * L_ + (size_t)ch * CHUNK;
    float a0 = P[P_FBA + row * 4], a1 = P[P_FBA + row * 4 + 1];
    float a2 = P[P_FBA + row * 4 + 2], a3 = P[P_FBA + row * 4 + 3];
    float k0 = P[P_KFB + row * 2], k1 = P[P_KFB + row * 2 + 1];
    float v0 = 0, v1 = 0;
    for (int i = 0; i < CHUNK; ++i) {
        float xv = src[i];
        float n0 = a0 * v0 + a1 * v1 + k0 * xv;
        float n1 = a2 * v0 + a3 * v1 + k1 * xv;
        v0 = n0; v1 = n1;
    }
    w[(size_t)tid * 2] = v0; w[(size_t)tid * 2 + 1] = v1;
}

__global__ void fb_carry_kernel(const float* __restrict__ P,
                                const float* __restrict__ w,
                                float* __restrict__ sin) {
    int r = threadIdx.x;
    if (r >= B_) return;
    float a0 = P[P_FBA_CH + r * 4], a1 = P[P_FBA_CH + r * 4 + 1];
    float a2 = P[P_FBA_CH + r * 4 + 2], a3 = P[P_FBA_CH + r * 4 + 3];
    float v0 = 0, v1 = 0;
    for (int j = 0; j < NCH; ++j) {
        size_t b = ((size_t)r * NCH + j) * 2;
        sin[b] = v0; sin[b + 1] = v1;
        float n0 = a0 * v0 + a1 * v1 + w[b];
        float n1 = a2 * v0 + a3 * v1 + w[b + 1];
        v0 = n0; v1 = n1;
    }
}

__global__ void __launch_bounds__(256)
fb_replay_combine_kernel(const float* __restrict__ y4, const float* __restrict__ P,
                         const float* __restrict__ sin, float* __restrict__ out) {
    int tid = blockIdx.x * blockDim.x + threadIdx.x;
    int row = tid / NCH, ch = tid % NCH;
    const float* src = y4  + (size_t)row * L_ + (size_t)ch * CHUNK;
    float*       dst = out + (size_t)row * L_ + (size_t)ch * CHUNK;
    float a0 = P[P_FBA + row * 4], a1 = P[P_FBA + row * 4 + 1];
    float a2 = P[P_FBA + row * 4 + 2], a3 = P[P_FBA + row * 4 + 3];
    float k0 = P[P_KFB + row * 2], k1 = P[P_KFB + row * 2 + 1];
    float b0 = P[P_FBB0 + row];
    float fbmix = P[P_FBMIX];
    float v0 = sin[(size_t)tid * 2], v1 = sin[(size_t)tid * 2 + 1];
    for (int i = 0; i < CHUNK; ++i) {
        float xv  = src[i];
        float yfb = b0 * xv + v0;
        dst[i] = xv - fbmix * yfb;
        float n0 = a0 * v0 + a1 * v1 + k0 * xv;
        float n1 = a2 * v0 + a3 * v1 + k1 * xv;
        v0 = n0; v1 = n1;
    }
}

// ---------------------------------------------------------------- launch
extern "C" void kernel_launch(void* const* d_in, const int* in_sizes, int n_in,
                              void* d_out, int out_size, void* d_ws, size_t ws_size,
                              hipStream_t stream) {
    const float* x    = (const float*)d_in[0];
    const float* kn   = (const float*)d_in[1];
    const float* env  = (const float*)d_in[2];
    const float* preR = (const float*)d_in[3];
    const float* posR = (const float*)d_in[4];
    const float* gwi  = (const float*)d_in[5];
    const float* gwh  = (const float*)d_in[6];
    const float* gbi  = (const float*)d_in[7];
    const float* gbh  = (const float*)d_in[8];
    const float* gow  = (const float*)d_in[9];
    const float* gob  = (const float*)d_in[10];
    const float* sw1  = (const float*)d_in[11];
    const float* sb1  = (const float*)d_in[12];
    const float* sw2  = (const float*)d_in[13];
    const float* sb2  = (const float*)d_in[14];
    const float* fw1  = (const float*)d_in[15];
    const float* fb1  = (const float*)d_in[16];
    const float* fw2  = (const float*)d_in[17];
    const float* fb2  = (const float*)d_in[18];
    const float* famt = (const float*)d_in[19];
    float* out = (float*)d_out;

    float* P         = (float*)d_ws;
    float* lastlocal = P + PARAM_FLOATS;        // 65536
    float* Ein       = lastlocal + NBLK_T;      // 65536
    float* wbuf      = Ein + NBLK_T;            // 16384
    float* sbuf      = wbuf + B_ * NCH * 4;     // 16384
    float* bufA      = sbuf + B_ * NCH * 4;     // NL_
    float* bufB      = bufA + NL_;              // NL_
    float* bufC      = bufB + NL_;              // NL_

    param_kernel<<<1, 1, 0, stream>>>(kn, env, preR, posR, sw1, sb1, sw2, sb2,
                                      fw1, fb1, fw2, fb2, famt, P);

    // envelope: WMMA local pass (4096 tiles, 8 waves/block), carry, apply+sag
    env_local_kernel<<<512, 256, 0, stream>>>(x, P, bufA, lastlocal);
    env_carry_kernel<<<1, 32, 0, stream>>>(P, lastlocal, Ein);
    sag_apply_kernel<<<NL_ / 256, 256, 0, stream>>>(x, P, bufA, Ein, bufB);

    // pre cascade: bufB -> bufC
    cascade_w_kernel<<<16, 256, 0, stream>>>(bufB, P + P_APRE, P + P_KPRE, wbuf);
    cascade_carry_kernel<<<1, 32, 0, stream>>>(P + P_APRE_CH, wbuf, sbuf);
    cascade_replay_kernel<<<16, 256, 0, stream>>>(bufB, P + P_APRE, P + P_KPRE,
                                                  P + P_GPRE, sbuf, bufC);
    // GRU: bufC -> bufA   (TDM double-buffered LDS staging)
    gru_kernel<<<1, 256, 0, stream>>>(bufC, gwi, gwh, gbi, gbh, gow, gob, bufA);

    // post cascade: bufA -> bufB (y4)
    cascade_w_kernel<<<16, 256, 0, stream>>>(bufA, P + P_APOST, P + P_KPOST, wbuf);
    cascade_carry_kernel<<<1, 32, 0, stream>>>(P + P_APOST_CH, wbuf, sbuf);
    cascade_replay_kernel<<<16, 256, 0, stream>>>(bufA, P + P_APOST, P + P_KPOST,
                                                  P + P_GPOST, sbuf, bufB);

    // feedback biquad on y4, fused final mix -> out
    fb_w_kernel<<<16, 256, 0, stream>>>(bufB, P, wbuf);
    fb_carry_kernel<<<1, 32, 0, stream>>>(P, wbuf, sbuf);
    fb_replay_combine_kernel<<<16, 256, 0, stream>>>(bufB, P, sbuf, out);
}
